// HMP_SphereNetModel_77017353552155
// MI455X (gfx1250) — matrix operand order repaired and checked by reading, not compiled, not run on hardware
//
#include <hip/hip_runtime.h>
#include <math.h>

// ---------------- types ----------------
typedef __attribute__((ext_vector_type(16))) __bf16 v16bf;
typedef __attribute__((ext_vector_type(8)))  float  v8f;
typedef __attribute__((ext_vector_type(4)))  float  f4;

static inline int cdiv(int a, int b) { return (a + b - 1) / b; }

#define N_EDGES 40000
#define N_NODES 5000
#define N_GRAPHS 16

__device__ __forceinline__ float fast_sigmoid(float x) {
  return __builtin_amdgcn_rcpf(1.f + __expf(-x));  // v_exp_f32 + v_rcp_f32
}

// ---------------- small utility kernels ----------------
__global__ void k_zero(float* p, int n) {
  int i = blockIdx.x * blockDim.x + threadIdx.x;
  if (i < n) p[i] = 0.f;
}

// Convert fp32 weight [K,N] -> bf16 in WMMA B-fragment swizzled order:
// block (kb = k/32, nb = n/16) stored as 32 lanes x 16 bf16:
//   lane = (n%16) + 16*(((k%32)/8)%2),  t = k%8 + 8*((k%32)>=16)
__global__ void k_cvt_swz(const float* __restrict__ src, __bf16* __restrict__ dst,
                          int K, int N) {
  int idx = blockIdx.x * blockDim.x + threadIdx.x;
  if (idx >= K * N) return;
  int k = idx / N, n = idx - k * N;
  int kb = k >> 5, kk = k & 31;
  int nb = n >> 4, nn = n & 15;
  int lane = nn + (((kk >> 3) & 1) << 4);
  int t = (kk & 7) + ((kk >> 4) << 3);
  dst[(((size_t)kb * (N >> 4) + nb) * 32 + lane) * 16 + t] = (__bf16)src[idx];
}

// out[I,J] = A[I,K] @ B[K,J]   (tiny fp32 matmul for combined small weights)
__global__ void k_ab(const float* __restrict__ A, const float* __restrict__ B,
                     float* __restrict__ out, int I, int K, int J) {
  int idx = blockIdx.x * blockDim.x + threadIdx.x;
  if (idx >= I * J) return;
  int i = idx / J, j = idx - i * J;
  float acc = 0.f;
  for (int k = 0; k < K; ++k) acc += A[i * K + k] * B[k * J + j];
  out[idx] = acc;
}

// ---------------- WMMA bf16 GEMM: C = [addend +] act(A@W [+ bias]) ----------
// A fp32 [M,K] row-major (rows clamped at boundary, stores guarded only in the
// boundary block). Wz bf16 fragment-swizzled (see k_cvt_swz). K%32==0, N%64==0.
// Block = 8 waves; each wave owns an independent 32x32 tile of a 128x64 block
// tile. No LDS, no barriers.
template <int ACT, int BIAS, int ADD, bool GUARD>
__device__ __forceinline__ void gemm_epilogue(const v8f (&c)[2][2],
                                              const float* __restrict__ bias,
                                              const float* __restrict__ addend,
                                              float* __restrict__ C, int M, int N,
                                              int bm, int bn, int lh, int hi) {
#pragma unroll
  for (int j = 0; j < 2; ++j) {
    int col = bn + j * 16 + lh;
    float bcol = BIAS ? bias[col] : 0.f;
#pragma unroll
    for (int i = 0; i < 2; ++i) {
      int tm = bm + i * 16;
#pragma unroll
      for (int r = 0; r < 8; ++r) {
        int row = tm + r + hi * 8;
        if (GUARD && row >= M) continue;
        float acc = c[i][j][r];
        if (BIAS) acc += bcol;
        if (ACT == 1) acc *= fast_sigmoid(acc);
        if (ADD) acc += addend[(size_t)row * N + col];
        C[(size_t)row * N + col] = acc;
      }
    }
  }
}

template <int ACT, int BIAS, int ADD>  // ACT: 0=none 1=swish
__global__ __launch_bounds__(256) void k_gemm(const float* __restrict__ A,
                                              const __bf16* __restrict__ Wz,
                                              const float* __restrict__ bias,
                                              const float* __restrict__ addend,
                                              float* __restrict__ C,
                                              int M, int N, int K) {
  int wave = threadIdx.x >> 5, lane = threadIdx.x & 31;
  int lh = lane & 15, hi = lane >> 4;
  int bm = blockIdx.x * 128 + (wave >> 1) * 32;
  int bn = blockIdx.y * 64 + (wave & 1) * 32;
  int ntiles = N >> 4;
  const v16bf* Wv = (const v16bf*)Wz;

  v8f c[2][2] = {};

  for (int k0 = 0; k0 < K; k0 += 32) {
    // A fragments: lane (lh,hi) needs K = {hi*8+0..7, hi*8+16..23} of its row.
    v16bf a[2];
#pragma unroll
    for (int i = 0; i < 2; ++i) {
      int row = bm + i * 16 + lh;
      row = (row < M) ? row : (M - 1);  // branchless boundary clamp
      const f4* ap = (const f4*)(A + (size_t)row * K + k0 + hi * 8);
      f4 f0 = ap[0], f1 = ap[1], f2 = ap[4], f3 = ap[5];
#pragma unroll
      for (int t = 0; t < 4; ++t) {
        a[i][t]      = (__bf16)f0[t];
        a[i][4 + t]  = (__bf16)f1[t];
        a[i][8 + t]  = (__bf16)f2[t];
        a[i][12 + t] = (__bf16)f3[t];
      }
    }
    // B fragments: direct 32-byte loads from swizzled weights.
    size_t wbase = ((size_t)(k0 >> 5) * ntiles + (bn >> 4)) * 32;
    v16bf b0 = Wv[wbase + lane];
    v16bf b1 = Wv[wbase + 32 + lane];

    c[0][0] = __builtin_amdgcn_wmma_f32_16x16x32_bf16(false, a[0], false, b0, (short)0, c[0][0], false, false);
    c[0][1] = __builtin_amdgcn_wmma_f32_16x16x32_bf16(false, a[0], false, b1, (short)0, c[0][1], false, false);
    c[1][0] = __builtin_amdgcn_wmma_f32_16x16x32_bf16(false, a[1], false, b0, (short)0, c[1][0], false, false);
    c[1][1] = __builtin_amdgcn_wmma_f32_16x16x32_bf16(false, a[1], false, b1, (short)0, c[1][1], false, false);
  }

  // Wave-uniform split: only the boundary block pays per-element guards.
  if (bm + 32 <= M)
    gemm_epilogue<ACT, BIAS, ADD, false>(c, bias, addend, C, M, N, bm, bn, lh, hi);
  else
    gemm_epilogue<ACT, BIAS, ADD, true>(c, bias, addend, C, M, N, bm, bn, lh, hi);
}

// ---------------- small-K contraction (K<=8) -------------------------------
// out[m,n] = act(sum_k A[m,k]*W[k,n]) [*mul[m,n]] [*rowscale[m]]
template <int ACT>
__global__ void k_smallk(const float* __restrict__ A, const float* __restrict__ W,
                         const float* __restrict__ mul, const float* __restrict__ rowscale,
                         float* __restrict__ out, int M, int N, int K) {
  int idx = blockIdx.x * blockDim.x + threadIdx.x;
  if (idx >= M * N) return;
  int m = idx / N, n = idx - m * N;
  float acc = 0.f;
  for (int k = 0; k < K; ++k) acc += A[(size_t)m * K + k] * W[k * N + n];
  if (ACT == 1) acc *= fast_sigmoid(acc);
  if (mul) acc *= mul[idx];
  if (rowscale) acc *= rowscale[m];
  out[idx] = acc;
}

// ---------------- geometry / basis ----------------
__global__ void k_edge_geom(const float* __restrict__ pos, const int* __restrict__ ei,
                            float* __restrict__ vec, float* __restrict__ dist,
                            float* __restrict__ rbf, int E) {
  int e = blockIdx.x * blockDim.x + threadIdx.x;
  if (e >= E) return;
  int j = ei[e], i = ei[E + e];
  float vx = pos[i * 3 + 0] - pos[j * 3 + 0];
  float vy = pos[i * 3 + 1] - pos[j * 3 + 1];
  float vz = pos[i * 3 + 2] - pos[j * 3 + 2];
  vec[e * 3 + 0] = vx; vec[e * 3 + 1] = vy; vec[e * 3 + 2] = vz;
  float d = sqrtf(vx * vx + vy * vy + vz * vz);
  dist[e] = d;
  float x = d * 0.2f;  // /CUTOFF
  float x2 = x * x;
  float x5 = x2 * x2 * x;
  float env = 1.f / x - 28.f * x5 + 48.f * x5 * x - 21.f * x5 * x2;  // p=6
  env = (x < 1.f) ? env : 0.f;
  for (int r = 0; r < 6; ++r)
    rbf[e * 6 + r] = env * sinf((float)(r + 1) * 3.14159265358979323846f * x);
}

__global__ void k_trip_geom(const float* __restrict__ vec, const int* __restrict__ kj,
                            const int* __restrict__ ji, float* __restrict__ angle,
                            float* __restrict__ torsion, int T) {
  int t = blockIdx.x * blockDim.x + threadIdx.x;
  if (t >= T) return;
  int eji = ji[t], ekj = kj[t];
  float ax = vec[eji * 3 + 0], ay = vec[eji * 3 + 1], az = vec[eji * 3 + 2];
  float bx = -vec[ekj * 3 + 0], by = -vec[ekj * 3 + 1], bz = -vec[ekj * 3 + 2];
  float dot = ax * bx + ay * by + az * bz;
  float cx = ay * bz - az * by;
  float cy = az * bx - ax * bz;
  float cz = ax * by - ay * bx;
  float cn = sqrtf(cx * cx + cy * cy + cz * cz);
  angle[t] = atan2f(cn + 1e-12f, dot);
  float m2x = cy * az - cz * ay;  // cross(crs, v_ji).x
  torsion[t] = atan2f(m2x, cx + 1e-9f);
}

// ---------------- triplet message + segment sum (wave-per-triplet) --------
// Lane n owns outputs n and n+32 of the 64-wide message:
//   agg[ji][n] += xkj64[kj][n] * (sum_q p_s[q]*Wsbf[q][n])
//                              * (sum_q p_t[q]*Wt[q][n]) [* edge_w[kj]*edge_w[ji]]
// Wsbf/Wt rows are read with lane-consecutive addresses (coalesced 128B), and
// the two atomics per lane are wave-coalesced global_atomic_add_f32.
__global__ __launch_bounds__(256) void k_trip_msg(
    const float* __restrict__ angle, const float* __restrict__ torsion,
    const float* __restrict__ rbf, const int* __restrict__ kj, const int* __restrict__ ji,
    const float* __restrict__ xkj, const float* __restrict__ Wsbf,
    const float* __restrict__ Wt, const float* __restrict__ edge_w,
    float* __restrict__ agg, int T) {
  int wid = blockIdx.x * (blockDim.x >> 5) + (threadIdx.x >> 5);
  if (wid >= T) return;
  int lane = threadIdx.x & 31;
  int ekj = kj[wid], eji = ji[wid];
  float tw = 1.f;
  if (edge_w) {  // wave-uniform early out in the masked pass
    tw = edge_w[ekj] * edge_w[eji];
    if (tw == 0.f) return;
  }
  float a = angle[wid], to = torsion[wid];
  float ca[7], ct[7], rad[6];
#pragma unroll
  for (int s = 0; s < 7; ++s) {
    ca[s] = cosf((float)s * a);
    ct[s] = cosf((float)s * to);
  }
#pragma unroll
  for (int r = 0; r < 6; ++r) rad[r] = rbf[ekj * 6 + r];

  float s1a = 0.f, s1b = 0.f, s2a = 0.f, s2b = 0.f;
#pragma unroll
  for (int s = 0; s < 7; ++s)
#pragma unroll
    for (int r = 0; r < 6; ++r) {
      int q = s * 6 + r;
      float ps = ca[s] * rad[r];
      float pt = ct[s] * rad[r];
      s1a += ps * Wsbf[q * 64 + lane];
      s1b += ps * Wsbf[q * 64 + 32 + lane];
      s2a += pt * Wt[q * 64 + lane];
      s2b += pt * Wt[q * 64 + 32 + lane];
    }
  const float* xk = xkj + (size_t)ekj * 64;
  float* ag = agg + (size_t)eji * 64;
  atomicAdd(&ag[lane],      xk[lane]      * s1a * s2a * tw);
  atomicAdd(&ag[32 + lane], xk[32 + lane] * s1b * s2b * tw);
}

// ---------------- scatter add rows (segment sum) ----------------
__global__ void k_scatter_add(const float* __restrict__ src, const int* __restrict__ idx,
                              float* __restrict__ dst, int R, int C) {
  int g = blockIdx.x * blockDim.x + threadIdx.x;
  if (g >= R * C) return;
  int r = g / C, c = g - r * C;
  atomicAdd(&dst[(size_t)idx[r] * C + c], src[g]);
}

// ---------------- init_e constant vector ----------------
// cvec[n] = b_cat[n] + sum_k emb0[k]*(Wcat[k,n] + Wcat[128+k,n])
__global__ void k_cvec(const float* __restrict__ emb, const float* __restrict__ Wcat,
                       const float* __restrict__ bcat, float* __restrict__ cvec) {
  int n = threadIdx.x;
  float acc = bcat[n];
  for (int k = 0; k < 128; ++k)
    acc += emb[k] * (Wcat[k * 128 + n] + Wcat[(128 + k) * 128 + n]);
  cvec[n] = acc;
}

// ---------------- mask MLP: 16 -> 32 -> sigmoid ----------------
__global__ void k_mask(const float* __restrict__ vloc, const float* __restrict__ W1,
                       const float* __restrict__ b1, const float* __restrict__ W2,
                       const float* __restrict__ b2, float* __restrict__ m,
                       float* __restrict__ mh, int Nn) {
  int n = blockIdx.x * blockDim.x + threadIdx.x;
  if (n >= Nn) return;
  float s[16];
#pragma unroll
  for (int i = 0; i < 16; ++i) s[i] = vloc[(size_t)n * 128 + i];
  float acc2 = b2[0];
  for (int h = 0; h < 32; ++h) {
    float a = b1[h];
#pragma unroll
    for (int i = 0; i < 16; ++i) a += s[i] * W1[i * 32 + h];
    a = fmaxf(a, 0.f);
    acc2 += a * W2[h];
  }
  float mm = fast_sigmoid(acc2);
  m[n] = mm;
  mh[n] = (mm > 0.5f) ? 1.f : 0.f;
}

__global__ void k_edge_w(const int* __restrict__ ei, const float* __restrict__ mh,
                         float* __restrict__ ew, int E) {
  int e = blockIdx.x * blockDim.x + threadIdx.x;
  if (e >= E) return;
  ew[e] = mh[ei[e]] * mh[ei[E + e]];
}

// v = (1-m)*v_local + m*mh*(v_local + u2)
__global__ void k_combine(const float* __restrict__ vloc, const float* __restrict__ u2,
                          const float* __restrict__ m, const float* __restrict__ mh,
                          float* __restrict__ v, int Nn) {
  int g = blockIdx.x * blockDim.x + threadIdx.x;
  if (g >= Nn * 128) return;
  int n = g >> 7;
  float mm = m[n], h = mh[n];
  float vl = vloc[g];
  v[g] = (1.f - mm) * vl + mm * h * (vl + u2[g]);
}

// ---------------- pred head: relu(pooled@W1+b1)@W2 + b2 ----------------
__global__ void k_pred(const float* __restrict__ pooled, const float* __restrict__ W1,
                       const float* __restrict__ b1, const float* __restrict__ W2,
                       const float* __restrict__ b2, float* __restrict__ out) {
  __shared__ float hid[128];
  int g = blockIdx.x, t = threadIdx.x;
  float acc = b1[t];
  for (int k = 0; k < 128; ++k) acc += pooled[g * 128 + k] * W1[k * 128 + t];
  hid[t] = fmaxf(acc, 0.f);
  __syncthreads();
  if (t == 0) {
    float o = b2[0];
    for (int k = 0; k < 128; ++k) o += hid[k] * W2[k];
    out[g] = o;
  }
}

// ---------------- host orchestration ----------------
struct UE {
  const __bf16 *Wji, *Wkj, *Wdown, *Wup, *Wskip, *bW1, *bW2;
  const __bf16 *aW1[2], *aW2[2];
};
struct UV {
  const __bf16 *Wup, *lw[3], *Wout;
};

static void gemm_launch(hipStream_t s, const float* A, const __bf16* W, const float* bias,
                        const float* add, float* C, int M, int N, int K, int act) {
  dim3 g(cdiv(M, 128), N / 64), b(256);
  if (act == 1) {
    if (bias) {
      if (add) k_gemm<1, 1, 1><<<g, b, 0, s>>>(A, W, bias, add, C, M, N, K);
      else     k_gemm<1, 1, 0><<<g, b, 0, s>>>(A, W, bias, add, C, M, N, K);
    } else {
      if (add) k_gemm<1, 0, 1><<<g, b, 0, s>>>(A, W, bias, add, C, M, N, K);
      else     k_gemm<1, 0, 0><<<g, b, 0, s>>>(A, W, bias, add, C, M, N, K);
    }
  } else {
    if (add)   k_gemm<0, 0, 1><<<g, b, 0, s>>>(A, W, bias, add, C, M, N, K);
    else       k_gemm<0, 0, 0><<<g, b, 0, s>>>(A, W, bias, add, C, M, N, K);
  }
}
static void smallk_launch(hipStream_t s, const float* A, const float* W, const float* mul,
                          const float* rs, float* out, int M, int N, int K, int act) {
  dim3 g(cdiv(M * N, 256)), b(256);
  if (act == 1) k_smallk<1><<<g, b, 0, s>>>(A, W, mul, rs, out, M, N, K);
  else          k_smallk<0><<<g, b, 0, s>>>(A, W, mul, rs, out, M, N, K);
}

// Param leaf indexing (jax pytree order: dict keys sorted, lists/tuples in order):
//  0 z, 1 pos, 2 edge_index[2,E], 3 batch, 4 idx_kj, 5 idx_ji
//  init_e: 6 W_cat(384,128) 7 W_rbf0(6,128) 8 W_rbf1(6,128) 9 b_cat 10 emb(1,128)
//  init_v: 11 W_out(256,128) 12 W_up(128,256) 13 b_up 14..19 [W,b]x3 lins(256,256)
//  layers base 20, stride 36:
//   ue: +0 W_down(128,64) +1 W_ji +2 W_kj +3 W_rbf1(6,8) +4 W_rbf2(8,128)
//       +5 W_rbf_out(6,128) +6 W_sbf1(42,8) +7 W_sbf2(8,64) +8 W_skip
//       +9 W_t1(42,8) +10 W_t2(8,64) +11 W_up(64,128)
//       +12..19 after[0..1]{W1,b1,W2,b2} +20 b_ji +21 b_kj +22 b_skip
//       +23..26 before{W1,b1,W2,b2}
//   uv: +27 W_out +28 W_up +29 b_up +30..35 [W,b]x3
//  ms base 128 stride 4: W1(16,32) W2(32,1) b1 b2
//  pred base 140: W1(128,128) W2(128,1) b1 b2
extern "C" void kernel_launch(void* const* d_in, const int* in_sizes, int n_in,
                              void* d_out, int out_size, void* d_ws, size_t ws_size,
                              hipStream_t stream) {
  const int E = N_EDGES, Nn = N_NODES, G = N_GRAPHS;
  const int T = in_sizes[4];

  const float* pos = (const float*)d_in[1];
  const int* ei = (const int*)d_in[2];
  const int* batch = (const int*)d_in[3];
  const int* kj = (const int*)d_in[4];
  const int* ji = (const int*)d_in[5];
  const int* i_idx = ei + E;
  auto F = [&](int i) { return (const float*)d_in[i]; };

  // ---- workspace bump allocator ----
  char* wsp = (char*)d_ws;
  size_t off = 0;
  auto alloc = [&](size_t bytes) -> char* {
    char* p = wsp + off;
    off += (bytes + 255) & ~(size_t)255;
    return p;
  };
  float* vec = (float*)alloc((size_t)E * 3 * 4);
  float* dist = (float*)alloc((size_t)E * 4);
  float* rbf = (float*)alloc((size_t)E * 6 * 4);
  float* ang = (float*)alloc((size_t)T * 4);
  float* tor = (float*)alloc((size_t)T * 4);
  float* B0 = (float*)alloc((size_t)E * 128 * 4);  // e1 / h1
  float* B1 = (float*)alloc((size_t)E * 128 * 4);  // x_ji / e2 / h(call2)
  float* B2 = (float*)alloc((size_t)E * 128 * 4);  // x_kj / tmp
  float* B3 = (float*)alloc((size_t)E * 128 * 4);  // h / e2m
  float* xkj64 = (float*)alloc((size_t)E * 64 * 4);
  float* agg = (float*)alloc((size_t)E * 64 * 4);
  float* ew = (float*)alloc((size_t)E * 4);
  float* vsum = (float*)alloc((size_t)Nn * 128 * 4);
  float* vA = (float*)alloc((size_t)Nn * 256 * 4);
  float* vB = (float*)alloc((size_t)Nn * 256 * 4);
  float* vcur = (float*)alloc((size_t)Nn * 128 * 4);
  float* vloc = (float*)alloc((size_t)Nn * 128 * 4);
  float* u2 = (float*)alloc((size_t)Nn * 128 * 4);
  float* mbuf = (float*)alloc((size_t)Nn * 4);
  float* mhbuf = (float*)alloc((size_t)Nn * 4);
  float* pooled = (float*)alloc((size_t)G * 128 * 4);
  float* cvec = (float*)alloc(128 * 4);
  float* Wrbf12[3];
  float* Wsbf[3];
  float* Wt[3];
  for (int l = 0; l < 3; ++l) {
    Wrbf12[l] = (float*)alloc(6 * 128 * 4);
    Wsbf[l] = (float*)alloc(42 * 64 * 4);
    Wt[l] = (float*)alloc(42 * 64 * 4);
  }
  char* barena = alloc((size_t)8 << 20);
  if (off > ws_size) return;  // not enough scratch; refuse to scribble

  size_t boff = 0;
  auto cvtW = [&](const float* src, int K, int N) -> const __bf16* {
    __bf16* dst = (__bf16*)(barena + boff);
    int elems = K * N;
    boff += ((size_t)elems * 2 + 255) & ~(size_t)255;
    k_cvt_swz<<<cdiv(elems, 256), 256, 0, stream>>>(src, dst, K, N);
    return dst;
  };
  auto zero = [&](float* p, int n) { k_zero<<<cdiv(n, 256), 256, 0, stream>>>(p, n); };

  // ---- convert weights to bf16 (fragment-swizzled), combined small weights ----
  const __bf16* Wcat_bf = cvtW(F(6) + 256 * 128, 128, 128);  // rbf0 block of W_cat
  UV uvinit;
  uvinit.Wup = cvtW(F(12), 128, 256);
  uvinit.lw[0] = cvtW(F(14), 256, 256);
  uvinit.lw[1] = cvtW(F(16), 256, 256);
  uvinit.lw[2] = cvtW(F(18), 256, 256);
  uvinit.Wout = cvtW(F(11), 256, 128);
  UE ue[3];
  UV uv[3];
  for (int l = 0; l < 3; ++l) {
    int b = 20 + 36 * l, u = b + 27;
    ue[l].Wji = cvtW(F(b + 1), 128, 128);
    ue[l].Wkj = cvtW(F(b + 2), 128, 128);
    ue[l].Wdown = cvtW(F(b + 0), 128, 64);
    ue[l].Wup = cvtW(F(b + 11), 64, 128);
    ue[l].Wskip = cvtW(F(b + 8), 128, 128);
    ue[l].bW1 = cvtW(F(b + 23), 128, 128);
    ue[l].bW2 = cvtW(F(b + 25), 128, 128);
    ue[l].aW1[0] = cvtW(F(b + 12), 128, 128);
    ue[l].aW2[0] = cvtW(F(b + 14), 128, 128);
    ue[l].aW1[1] = cvtW(F(b + 16), 128, 128);
    ue[l].aW2[1] = cvtW(F(b + 18), 128, 128);
    uv[l].Wup = cvtW(F(u + 1), 128, 256);
    uv[l].lw[0] = cvtW(F(u + 3), 256, 256);
    uv[l].lw[1] = cvtW(F(u + 5), 256, 256);
    uv[l].lw[2] = cvtW(F(u + 7), 256, 256);
    uv[l].Wout = cvtW(F(u + 0), 256, 128);
    k_ab<<<cdiv(6 * 128, 256), 256, 0, stream>>>(F(b + 3), F(b + 4), Wrbf12[l], 6, 8, 128);
    k_ab<<<cdiv(42 * 64, 256), 256, 0, stream>>>(F(b + 6), F(b + 7), Wsbf[l], 42, 8, 64);
    k_ab<<<cdiv(42 * 64, 256), 256, 0, stream>>>(F(b + 9), F(b + 10), Wt[l], 42, 8, 64);
  }

  // ---- geometry + basis ----
  k_edge_geom<<<cdiv(E, 256), 256, 0, stream>>>(pos, ei, vec, dist, rbf, E);
  k_trip_geom<<<cdiv(T, 256), 256, 0, stream>>>(vec, kj, ji, ang, tor, T);

  // ---- init_e (z indexes 1-row emb -> constant contribution) ----
  k_cvec<<<1, 128, 0, stream>>>(F(10), F(6), F(9), cvec);
  smallk_launch(stream, rbf, F(7), nullptr, nullptr, B2, E, 128, 6, 1);         // rbf0
  gemm_launch(stream, B2, Wcat_bf, cvec, nullptr, B0, E, 128, 128, 1);          // e1
  smallk_launch(stream, rbf, F(8), B0, nullptr, B1, E, 128, 6, 0);              // e2

  auto run_update_v = [&](const UV& w, const float* bup, const float* lb0, const float* lb1,
                          const float* lb2, const float* e2, const float* addend, float* outv) {
    zero(vsum, Nn * 128);
    k_scatter_add<<<cdiv(E * 128, 256), 256, 0, stream>>>(e2, i_idx, vsum, E, 128);
    gemm_launch(stream, vsum, w.Wup, bup, nullptr, vA, Nn, 256, 128, 1);
    gemm_launch(stream, vA, w.lw[0], lb0, nullptr, vB, Nn, 256, 256, 1);
    gemm_launch(stream, vB, w.lw[1], lb1, nullptr, vA, Nn, 256, 256, 1);
    gemm_launch(stream, vA, w.lw[2], lb2, nullptr, vB, Nn, 256, 256, 1);
    gemm_launch(stream, vB, w.Wout, nullptr, addend, outv, Nn, 128, 256, 0);
  };

  auto run_update_e = [&](int l, const float* e1in, float* hdst, float* e2dst,
                          const float* ewptr) {
    int b = 20 + 36 * l;
    const UE& w = ue[l];
    gemm_launch(stream, e1in, w.Wji, F(b + 20), nullptr, B1, E, 128, 128, 1);    // x_ji
    gemm_launch(stream, e1in, w.Wkj, F(b + 21), nullptr, B2, E, 128, 128, 1);    // x_kj
    smallk_launch(stream, rbf, Wrbf12[l], B2, nullptr, B2, E, 128, 6, 0);        // *= rbf proj
    gemm_launch(stream, B2, w.Wdown, nullptr, nullptr, xkj64, E, 64, 128, 1);    // down
    zero(agg, E * 64);
    k_trip_msg<<<cdiv(T, 8), 256, 0, stream>>>(ang, tor, rbf, kj, ji, xkj64,
                                               Wsbf[l], Wt[l], ewptr, agg, T);
    gemm_launch(stream, agg, w.Wup, nullptr, B1, B3, E, 128, 64, 1);             // h
    gemm_launch(stream, B3, w.bW1, F(b + 24), nullptr, B2, E, 128, 128, 1);      // before res
    gemm_launch(stream, B2, w.bW2, F(b + 26), B3, B3, E, 128, 128, 1);
    gemm_launch(stream, B3, w.Wskip, F(b + 22), e1in, hdst, E, 128, 128, 1);     // skip + e1
    for (int a = 0; a < 2; ++a) {                                                // after res x2
      gemm_launch(stream, hdst, w.aW1[a], F(b + 13 + 4 * a), nullptr, B2, E, 128, 128, 1);
      gemm_launch(stream, B2, w.aW2[a], F(b + 15 + 4 * a), hdst, hdst, E, 128, 128, 1);
    }
    smallk_launch(stream, rbf, F(b + 5), hdst, ewptr, e2dst, E, 128, 6, 0);      // e2
  };

  run_update_v(uvinit, F(13), F(15), F(17), F(19), B1, nullptr, vcur);           // v

  for (int l = 0; l < 3; ++l) {
    int u = 20 + 36 * l + 27;
    run_update_e(l, B0, B0, B1, nullptr);            // h1 -> B0, e2 -> B1
    run_update_v(uv[l], F(u + 2), F(u + 4), F(u + 6), F(u + 8), B1, vcur, vloc); // v_local
    int msb = 128 + 4 * l;
    k_mask<<<cdiv(Nn, 128), 128, 0, stream>>>(vloc, F(msb + 0), F(msb + 2),
                                              F(msb + 1), F(msb + 3), mbuf, mhbuf, Nn);
    k_edge_w<<<cdiv(E, 256), 256, 0, stream>>>(ei, mhbuf, ew, E);
    run_update_e(l, B0, B1, B3, ew);                 // masked pass: e2m -> B3 (B0=h1 kept)
    run_update_v(uv[l], F(u + 2), F(u + 4), F(u + 6), F(u + 8), B3, nullptr, u2);
    k_combine<<<cdiv(Nn * 128, 256), 256, 0, stream>>>(vloc, u2, mbuf, mhbuf, vcur, Nn);
    // e1 for next layer is h1, already in B0
  }

  zero(pooled, G * 128);
  k_scatter_add<<<cdiv(Nn * 128, 256), 256, 0, stream>>>(vcur, batch, pooled, Nn, 128);
  k_pred<<<G, 128, 0, stream>>>(pooled, F(140), F(142), F(141), F(143), (float*)d_out);
}
